// GraphDecoder_81518479278101
// MI455X (gfx1250) — compile-verified, matrix operand order checked by reading, not run on hardware
//
#include <hip/hip_runtime.h>

// ---------------- problem constants ----------------
#define BGR    128          // graphs
#define NPG    120          // nodes per graph
#define FEAT   128
#define HEADS  4
#define HF     (HEADS*FEAT) // 512
#define NTOT   (BGR*NPG)    // 15360
#define NEDGE  (NTOT*16)    // 245760
#define ETOT   (NEDGE+NTOT) // 261120 (edges + self loops)
#define K0PAD  256          // 248 padded to multiple of 32
#define NEG_SLOPE 0.2f

typedef __bf16 bf16;
typedef __attribute__((ext_vector_type(16))) __bf16 v16bf;
typedef __attribute__((ext_vector_type(8)))  __bf16 bf16x8;
typedef __attribute__((ext_vector_type(8)))  float  v8f;
typedef __attribute__((ext_vector_type(4)))  unsigned int u32x4;
typedef __attribute__((ext_vector_type(8)))  int i32x8;
typedef __attribute__((ext_vector_type(4)))  int i32x4;

#if defined(__has_builtin)
#if __has_builtin(__builtin_amdgcn_tensor_load_to_lds) && __has_builtin(__builtin_amdgcn_s_wait_tensorcnt)
#define USE_TDM 1
#endif
#endif

// ---------------- helpers ----------------
__device__ __forceinline__ unsigned f2ord(float f) {
    unsigned u = __float_as_uint(f);
    return (u & 0x80000000u) ? ~u : (u | 0x80000000u);
}
__device__ __forceinline__ float ord2f(unsigned o) {
    unsigned u = (o & 0x80000000u) ? (o & 0x7fffffffu) : ~o;
    return __uint_as_float(u);
}
__device__ __forceinline__ void edge_sd(const int* __restrict__ ei, int e, int& s, int& d) {
    if (e < NEDGE) { s = ei[e]; d = ei[NEDGE + e]; }
    else           { s = d = e - NEDGE; }   // appended self loops
}

// ---------------- WMMA bf16 GEMM: C[M,N] = A[M,K] @ Bt[N,K]^T ----------------
// Block tile: 128(M) x 64(N), 8 waves. Weight panel (64 x K bf16, <=64KB) staged in LDS
// once per block via the Tensor Data Mover (wave 0 issues TENSOR_LOAD_TO_LDS, waits on
// TENSORcnt, block barrier). Each wave computes a 16x64 strip with 4 accumulators so
// each A-fragment feeds 4 WMMAs. Grid tiles M,N exactly (M%128==0, N%64==0, K%32==0).
__global__ __launch_bounds__(256) void gemm_bf16_wmma(
        const bf16* __restrict__ A, const bf16* __restrict__ Bt,
        const float* __restrict__ bias, float* __restrict__ C,
        int M, int N, int K, int relu) {
    __shared__ bf16 bpanel[64 * 512];       // 64 KB (K <= 512)

    const int wid    = threadIdx.x >> 5;
    const int lane   = threadIdx.x & 31;
    const int tilesM = M >> 7;
    const int bm     = blockIdx.x % tilesM;
    const int bn     = blockIdx.x / tilesM;

#ifdef USE_TDM
    // ---- TDM stage of the 64-column weight panel: 1-D copy, data_size = 8 bytes ----
    if (wid == 0) {
        const unsigned elems8 = (unsigned)(64 * K) >> 2;       // (64*K*2)/8 qwords
        const unsigned long long ga = (unsigned long long)(const void*)(Bt + (size_t)bn * 64 * K);
        const unsigned lds = (unsigned)(unsigned long long)(const void*)bpanel; // addr[31:0] = LDS offset
        u32x4 g0;
        g0[0] = 1u;                                            // count=1, user descriptor
        g0[1] = lds;                                           // lds_addr [63:32]
        g0[2] = (unsigned)(ga & 0xffffffffu);                  // global_addr [95:64]
        g0[3] = ((unsigned)(ga >> 32) & 0x01ffffffu) | (2u << 30); // ga[56:32] | type=2
        i32x8 g1;
        g1[0] = (int)(3u << 16);                               // mask=0, data_size=8B
        g1[1] = (int)((elems8 & 0xffffu) << 16);               // tensor_dim0[15:0] @ [63:48]
        g1[2] = (int)(((elems8 >> 16) & 0xffffu) | (1u << 16)); // tensor_dim0[31:16], tensor_dim1=1
        g1[3] = (int)((elems8 & 0xffffu) << 16);               // tile_dim0 @ [127:112]
        g1[4] = 0;                                             // tile_dim1=0, tile_dim2=0
        g1[5] = (int)elems8;                                   // tensor_dim0_stride[31:0]
        g1[6] = 0;                                             // stride hi, dim1_stride lo
        g1[7] = 0;
        i32x4 gz = {0, 0, 0, 0};
#if __clang_major__ >= 23
        i32x8 gz8 = {0, 0, 0, 0, 0, 0, 0, 0};
        __builtin_amdgcn_tensor_load_to_lds(g0, g1, gz, gz, gz8, 0);
#else
        __builtin_amdgcn_tensor_load_to_lds(g0, g1, gz, gz, 0);
#endif
        __builtin_amdgcn_s_wait_tensorcnt(0);
    }
    __syncthreads();
#else
    // ---- fallback: cooperative vector copy ----
    {
        const uint4* __restrict__ src = (const uint4*)(Bt + (size_t)bn * 64 * K);
        uint4* dst = (uint4*)bpanel;
        const int chunks = (64 * K) >> 3;   // 16-byte chunks
        for (int c = threadIdx.x; c < chunks; c += 256) dst[c] = src[c];
    }
    __syncthreads();
#endif

    const int halfsel = lane >> 4;          // 0: lanes 0-15, 1: lanes 16-31
    const int koff    = halfsel << 3;       // 0 or 8
    const int nlane   = lane & 15;
    const bf16* __restrict__ arow = A + (size_t)((bm << 7) + (wid << 4) + nlane) * K;

    v8f acc[4] = {};
    for (int ks = 0; ks < K; ks += 32) {
        bf16x8 a0 = *(const bf16x8*)(arow + ks + koff);
        bf16x8 a1 = *(const bf16x8*)(arow + ks + koff + 16);
        v16bf av;
#pragma unroll
        for (int i = 0; i < 8; ++i) { av[i] = a0[i]; av[8 + i] = a1[i]; }
#pragma unroll
        for (int ns = 0; ns < 4; ++ns) {
            const bf16* brow = bpanel + (size_t)((ns << 4) + nlane) * K;
            bf16x8 b0 = *(const bf16x8*)(brow + ks + koff);
            bf16x8 b1 = *(const bf16x8*)(brow + ks + koff + 16);
            v16bf bv;
#pragma unroll
            for (int i = 0; i < 8; ++i) { bv[i] = b0[i]; bv[8 + i] = b1[i]; }
            acc[ns] = __builtin_amdgcn_wmma_f32_16x16x32_bf16(
                          false, av, false, bv, (short)0, acc[ns], false, false);
        }
    }

    // ---- epilogue: optional bias + ReLU, fp32 store ----
    const int crow0 = (bm << 7) + (wid << 4) + (halfsel << 3);
#pragma unroll
    for (int ns = 0; ns < 4; ++ns) {
        const int ccol = (bn << 6) + (ns << 4) + nlane;
        float badd = bias ? bias[ccol] : 0.f;
#pragma unroll
        for (int v = 0; v < 8; ++v) {
            float x = acc[ns][v] + badd;
            if (relu) x = fmaxf(x, 0.f);
            C[(size_t)(crow0 + v) * N + ccol] = x;
        }
    }
}

// ---------------- weight convert: W[Kin,N] fp32 -> Wt[N,Kpad] bf16 (zero pad) --------
__global__ void wt_convert(const float* __restrict__ W, bf16* __restrict__ Wt,
                           int Kin, int Kpad, int N) {
    int i = blockIdx.x * blockDim.x + threadIdx.x;
    if (i >= N * Kpad) return;
    int n = i / Kpad, k = i - n * Kpad;
    float v = (k < Kin) ? W[(size_t)k * N + n] : 0.f;
    Wt[i] = (bf16)v;
}

// ---------------- init linear: hinit[b,j] = cat(z,cond)[b,:] @ w_init + b_init -------
__global__ void init_linear(const float* __restrict__ z, const float* __restrict__ cond,
                            const float* __restrict__ w, const float* __restrict__ b,
                            float* __restrict__ hinit) {
    int i = blockIdx.x * blockDim.x + threadIdx.x;
    if (i >= BGR * FEAT) return;
    int bi = i / FEAT, j = i - bi * FEAT;
    float acc = b[j];
    const float* zr = z + bi * 256;
    for (int k = 0; k < 256; ++k) acc += zr[k] * w[k * FEAT + j];
    const float* cr = cond + bi * 64;
    for (int k = 0; k < 64; ++k)  acc += cr[k] * w[(256 + k) * FEAT + j];
    hinit[i] = acc;
}

// ---------------- x0[n,k]: relu(hinit[batch]) || one_hot(order) || pad ---------------
__global__ void build_x0(const float* __restrict__ hinit, const int* __restrict__ batch,
                         bf16* __restrict__ x0) {
    int i = blockIdx.x * blockDim.x + threadIdx.x;
    if (i >= NTOT * K0PAD) return;
    int n = i >> 8, k = i & 255;
    int g = batch[n];
    float v = 0.f;
    if (k < FEAT) {
        v = fmaxf(hinit[g * FEAT + k], 0.f);
    } else if (k < FEAT + NPG) {
        int ord = n - g * NPG;
        v = (k - FEAT == ord) ? 1.f : 0.f;
    }
    x0[i] = (bf16)v;
}

// ---------------- per-node attention dots + reset of m/denom -------------------------
__global__ void node_attn(const float* __restrict__ h, const float* __restrict__ asrc,
                          const float* __restrict__ adst, float* __restrict__ es,
                          float* __restrict__ ed, unsigned* __restrict__ mEnc,
                          float* __restrict__ denom) {
    int i = blockIdx.x * blockDim.x + threadIdx.x;
    if (i >= NTOT * HEADS) return;
    int n = i >> 2, hd = i & 3;
    const float* hv = h + (size_t)n * HF + hd * FEAT;
    const float* as = asrc + hd * FEAT;
    const float* ad = adst + hd * FEAT;
    float s = 0.f, d = 0.f;
    for (int c = 0; c < FEAT; ++c) { float x = hv[c]; s += x * as[c]; d += x * ad[c]; }
    es[i] = s; ed[i] = d;
    mEnc[i] = f2ord(-INFINITY);
    denom[i] = 0.f;
}

// ---------------- edge pass 1: e = leaky(es[s]+ed[d]); segment max over dst ----------
__global__ void edge_max(const int* __restrict__ ei, const float* __restrict__ es,
                         const float* __restrict__ ed, float* __restrict__ pbuf,
                         unsigned* __restrict__ mEnc) {
    int i = blockIdx.x * blockDim.x + threadIdx.x;
    if (i >= ETOT * HEADS) return;
    int e = i >> 2, hd = i & 3;
    int s, d; edge_sd(ei, e, s, d);
    float v = es[s * HEADS + hd] + ed[d * HEADS + hd];
    v = (v > 0.f) ? v : NEG_SLOPE * v;
    pbuf[i] = v;
    atomicMax(&mEnc[d * HEADS + hd], f2ord(v));
}

// ---------------- edge pass 2: p = exp(e - m[dst]); segment sum ----------------------
__global__ void edge_exp(const int* __restrict__ ei, const unsigned* __restrict__ mEnc,
                         float* __restrict__ pbuf, float* __restrict__ denom) {
    int i = blockIdx.x * blockDim.x + threadIdx.x;
    if (i >= ETOT * HEADS) return;
    int e = i >> 2, hd = i & 3;
    int s, d; edge_sd(ei, e, s, d);
    float p = __expf(pbuf[i] - ord2f(mEnc[d * HEADS + hd]));
    pbuf[i] = p;
    atomicAdd(&denom[d * HEADS + hd], p);
}

// ---------------- zero fp32 buffer ---------------------------------------------------
__global__ void zero_f32(float* __restrict__ p, int n) {
    int i = blockIdx.x * blockDim.x + threadIdx.x;
    if (i < n) p[i] = 0.f;
}

// ---------------- edge pass 3: agg[dst] += h[src] * alpha ----------------------------
__global__ void edge_aggregate(const int* __restrict__ ei, const float* __restrict__ pbuf,
                               const float* __restrict__ denom, const float* __restrict__ h,
                               float* __restrict__ agg) {
    int i = blockIdx.x * blockDim.x + threadIdx.x;
    if (i >= ETOT * HEADS) return;
    int e = i >> 2, hd = i & 3;
    int s, d; edge_sd(ei, e, s, d);
    float coeff = pbuf[i] / denom[d * HEADS + hd];
    const float4* __restrict__ src = (const float4*)(h + (size_t)s * HF + hd * FEAT);
    float* __restrict__ dst = agg + (size_t)d * HF + hd * FEAT;
#pragma unroll 4
    for (int c = 0; c < FEAT / 4; ++c) {
        float4 v = src[c];
        atomicAdd(dst + 4 * c + 0, v.x * coeff);
        atomicAdd(dst + 4 * c + 1, v.y * coeff);
        atomicAdd(dst + 4 * c + 2, v.z * coeff);
        atomicAdd(dst + 4 * c + 3, v.w * coeff);
    }
}

// ---------------- epilogue: x_next = bf16(relu(agg + bias)) --------------------------
__global__ void bias_relu_bf16(const float* __restrict__ agg, const float* __restrict__ b,
                               bf16* __restrict__ xo, int cols) {
    int i = blockIdx.x * blockDim.x + threadIdx.x;
    if (i >= NTOT * cols) return;
    int j = i % cols;
    xo[i] = (bf16)fmaxf(agg[i] + b[j], 0.f);
}

// ---------------- final [N,128] @ [128,2] projections --------------------------------
__global__ void final2(const float* __restrict__ t, const float* __restrict__ w,
                       const float* __restrict__ b, float* __restrict__ out) {
    int i = blockIdx.x * blockDim.x + threadIdx.x;
    if (i >= NTOT * 2) return;
    int n = i >> 1, o = i & 1;
    float acc = b[o];
    const float* tr = t + (size_t)n * FEAT;
    for (int k = 0; k < FEAT; ++k) acc += tr[k] * w[k * 2 + o];
    out[i] = acc;
}

// ============================ host side ==============================================
static inline size_t alignup(size_t x) { return (x + 255) & ~(size_t)255; }

extern "C" void kernel_launch(void* const* d_in, const int* in_sizes, int n_in,
                              void* d_out, int out_size, void* d_ws, size_t ws_size,
                              hipStream_t stream) {
    (void)in_sizes; (void)n_in; (void)out_size; (void)ws_size;
    const float* z        = (const float*)d_in[0];
    const float* cond     = (const float*)d_in[1];
    const int*   ei       = (const int*)  d_in[2];
    const int*   batch    = (const int*)  d_in[3];
    const float* w_init   = (const float*)d_in[4];
    const float* b_init   = (const float*)d_in[5];
    const float* w_conv[3]   = {(const float*)d_in[6],  (const float*)d_in[10], (const float*)d_in[14]};
    const float* att_src[3]  = {(const float*)d_in[7],  (const float*)d_in[11], (const float*)d_in[15]};
    const float* att_dst[3]  = {(const float*)d_in[8],  (const float*)d_in[12], (const float*)d_in[16]};
    const float* b_conv[3]   = {(const float*)d_in[9],  (const float*)d_in[13], (const float*)d_in[17]};
    const float* w_dpos  = (const float*)d_in[18];
    const float* b_dpos  = (const float*)d_in[19];
    const float* w_fpos  = (const float*)d_in[20];
    const float* b_fpos  = (const float*)d_in[21];
    const float* w_dsize = (const float*)d_in[22];
    const float* b_dsize = (const float*)d_in[23];
    const float* w_fsize = (const float*)d_in[24];
    const float* b_fsize = (const float*)d_in[25];

    // ---- workspace arena ----
    char* ws = (char*)d_ws;
    size_t off = 0;
    auto take = [&](size_t bytes) { char* p = ws + off; off += alignup(bytes); return p; };
    bf16*     W0t   = (bf16*)    take((size_t)HF * K0PAD * 2);
    bf16*     W1t   = (bf16*)    take((size_t)HF * HF * 2);
    bf16*     W2t   = (bf16*)    take((size_t)HF * HF * 2);
    bf16*     Wpt   = (bf16*)    take((size_t)FEAT * HF * 2);
    bf16*     Wst   = (bf16*)    take((size_t)FEAT * HF * 2);
    float*    hinit = (float*)   take((size_t)BGR * FEAT * 4);
    bf16*     XA    = (bf16*)    take((size_t)NTOT * HF * 2);
    bf16*     XB    = (bf16*)    take((size_t)NTOT * HF * 2);
    float*    Hf    = (float*)   take((size_t)NTOT * HF * 4);
    float*    AGG   = (float*)   take((size_t)NTOT * HF * 4);
    float*    ES    = (float*)   take((size_t)NTOT * HEADS * 4);
    float*    ED    = (float*)   take((size_t)NTOT * HEADS * 4);
    unsigned* MENC  = (unsigned*)take((size_t)NTOT * HEADS * 4);
    float*    DEN   = (float*)   take((size_t)NTOT * HEADS * 4);
    float*    PBUF  = (float*)   take((size_t)ETOT * HEADS * 4);

    const int T = 256;
    auto blk = [](long n, int t) { return (int)((n + t - 1) / t); };

    // ---- weight conversion (transposed bf16, K padded) ----
    wt_convert<<<blk((long)HF * K0PAD, T), T, 0, stream>>>(w_conv[0], W0t, 248, K0PAD, HF);
    wt_convert<<<blk((long)HF * HF,   T), T, 0, stream>>>(w_conv[1], W1t, HF, HF, HF);
    wt_convert<<<blk((long)HF * HF,   T), T, 0, stream>>>(w_conv[2], W2t, HF, HF, HF);
    wt_convert<<<blk((long)FEAT * HF, T), T, 0, stream>>>(w_dpos,  Wpt, HF, HF, FEAT);
    wt_convert<<<blk((long)FEAT * HF, T), T, 0, stream>>>(w_dsize, Wst, HF, HF, FEAT);

    // ---- init feature + x0 ----
    init_linear<<<blk((long)BGR * FEAT, T), T, 0, stream>>>(z, cond, w_init, b_init, hinit);
    build_x0<<<blk((long)NTOT * K0PAD, T), T, 0, stream>>>(hinit, batch, XA);

    // ---- 3 GAT layers ----
    const bf16* Wt_l[3]  = {W0t, W1t, W2t};
    const int   K_l[3]   = {K0PAD, HF, HF};
    bf16*       xin_l[3] = {XA, XB, XA};
    bf16*       xout_l[3]= {XB, XA, XB};
    for (int l = 0; l < 3; ++l) {
        int gblocks = (NTOT / 128) * (HF / 64);   // exact tiling
        gemm_bf16_wmma<<<gblocks, 256, 0, stream>>>(
            xin_l[l], Wt_l[l], nullptr, Hf, NTOT, HF, K_l[l], 0);
        node_attn<<<blk((long)NTOT * HEADS, T), T, 0, stream>>>(
            Hf, att_src[l], att_dst[l], ES, ED, MENC, DEN);
        zero_f32<<<blk((long)NTOT * HF, T), T, 0, stream>>>(AGG, NTOT * HF);
        edge_max<<<blk((long)ETOT * HEADS, T), T, 0, stream>>>(ei, ES, ED, PBUF, MENC);
        edge_exp<<<blk((long)ETOT * HEADS, T), T, 0, stream>>>(ei, MENC, PBUF, DEN);
        edge_aggregate<<<blk((long)ETOT * HEADS, T), T, 0, stream>>>(ei, PBUF, DEN, Hf, AGG);
        bias_relu_bf16<<<blk((long)NTOT * HF, T), T, 0, stream>>>(AGG, b_conv[l], xout_l[l], HF);
    }

    // ---- heads: pos then size (x3 lives in XB) ----
    float* out = (float*)d_out;
    int hblocks = (NTOT / 128) * (FEAT / 64);
    gemm_bf16_wmma<<<hblocks, 256, 0, stream>>>(XB, Wpt, b_dpos, Hf, NTOT, FEAT, HF, 1);
    final2<<<blk((long)NTOT * 2, T), T, 0, stream>>>(Hf, w_fpos, b_fpos, out);
    gemm_bf16_wmma<<<hblocks, 256, 0, stream>>>(XB, Wst, b_dsize, Hf, NTOT, FEAT, HF, 1);
    final2<<<blk((long)NTOT * 2, T), T, 0, stream>>>(Hf, w_fsize, b_fsize, out + (size_t)NTOT * 2);
}
